// MOE_46110768890299
// MI455X (gfx1250) — matrix operand order repaired and checked by reading, not compiled, not run on hardware
//
#include <hip/hip_runtime.h>

// ---------------- problem constants ----------------
constexpr int L_TOK = 8192;
constexpr int HD    = 2048;
constexpr int INTER = 1024;
constexpr int NE    = 8;

// ---------------- WMMA types ----------------
typedef __attribute__((ext_vector_type(16))) __bf16        v16bf;
typedef __attribute__((ext_vector_type(8)))  float         v8f;

union FragCvt { uint4 q[2]; v16bf b; };

static __device__ inline v8f zero8() {
    v8f z = {0.f,0.f,0.f,0.f,0.f,0.f,0.f,0.f};
    return z;
}

// fp32 -> bf16 (round to nearest even), bit-level
static __device__ inline unsigned short f2bf(float f) {
    unsigned int u = __float_as_uint(f);
    u = u + 0x7fffu + ((u >> 16) & 1u);
    return (unsigned short)(u >> 16);
}

// ---------------- tiling ----------------
constexpr int BM  = 64;   // block rows (tokens)
constexpr int BN  = 128;  // block cols
constexpr int BK  = 32;   // K chunk (one WMMA K)
constexpr int LDA = 40;   // sA stride (ushorts): 80B rows -> 16B-aligned b128 ops
constexpr int LDB = 40;   // transposed weight tile stride (80B rows, b128 frag reads)

// ---- A fragment: 16x32 bf16 from row-major LDS tile [m][k] (stride LDA)
// ISA 7.12.2: lane m = lane&15; half h = lane>>4 adds +8 to K base.
// uints {j+4h} j=0..3 and {8+j+4h} -> two aligned uint4 loads (ds_load_b128)
static __device__ inline v16bf load_a_frag(const unsigned short* sA, int mbase, int lane) {
    int m = mbase + (lane & 15);
    int h = lane >> 4;
    const uint4* row = (const uint4*)(sA + m * LDA);
    FragCvt f;
    f.q[0] = row[h];
    f.q[1] = row[2 + h];
    return f.b;
}

// ---- B fragment: 32x16 bf16 from transposed LDS tile BT[n][k] (stride LDB)
// lane n = lane&15; half h holds K=8h..8h+7 and K=16+8h..16+8h+7
static __device__ inline v16bf load_b_frag(const unsigned short* sBT, int nbase, int lane) {
    int n = nbase + (lane & 15);
    int h = lane >> 4;
    const uint4* row = (const uint4*)(sBT + n * LDB);
    FragCvt f;
    f.q[0] = row[h];
    f.q[1] = row[2 + h];
    return f.b;
}

// ---- CDNA5 async copy helpers (ASYNCcnt-tracked, LDS-direct) ----
static __device__ inline void async_load_b128(void* lds_dst, unsigned voff_bytes,
                                              const void* base) {
    unsigned lds = (unsigned)(size_t)lds_dst;
    unsigned long long sbase = (unsigned long long)(size_t)base;
    asm volatile("global_load_async_to_lds_b128 %0, %1, %2"
                 :: "v"(lds), "v"(voff_bytes), "s"(sbase)
                 : "memory");
}
static __device__ inline void wait_asynccnt0() {
    asm volatile("s_wait_asynccnt 0x0" ::: "memory");
}

// =====================================================================
// Kernel 1: zero output, convert x -> bf16, zero expert counters
// =====================================================================
__global__ __launch_bounds__(256)
void moe_prep(const float* __restrict__ x, unsigned short* __restrict__ Xbf,
              float* __restrict__ out, int* __restrict__ cnt, size_t n) {
    size_t stride = (size_t)gridDim.x * blockDim.x;
    for (size_t i = (size_t)blockIdx.x * blockDim.x + threadIdx.x; i < n; i += stride) {
        Xbf[i] = f2bf(x[i]);
        out[i] = 0.f;
    }
    if (blockIdx.x == 0 && threadIdx.x < NE) cnt[threadIdx.x] = 0;
}

// =====================================================================
// Kernel 2: router — one wave per token; softmax + top-2; compact lists
// =====================================================================
__global__ __launch_bounds__(256)
void moe_router(const float* __restrict__ x, const float* __restrict__ Wgate,
                const float* __restrict__ bgate, int* __restrict__ idxbuf,
                float* __restrict__ wbuf, int* __restrict__ cnt) {
    const int lane = threadIdx.x & 31;
    const int wave = threadIdx.x >> 5;
    const int t = blockIdx.x * 8 + wave;

    float acc[NE];
#pragma unroll
    for (int e = 0; e < NE; ++e) acc[e] = 0.f;

    const float* xr = x + (size_t)t * HD;
    for (int k = lane; k < HD; k += 32) {
        float xv = xr[k];
        const float* wr = Wgate + (size_t)k * NE;
#pragma unroll
        for (int e = 0; e < NE; ++e) acc[e] += xv * wr[e];
    }
#pragma unroll
    for (int e = 0; e < NE; ++e) {
#pragma unroll
        for (int off = 16; off > 0; off >>= 1)
            acc[e] += __shfl_xor(acc[e], off, 32);
    }
    if (lane == 0) {
        float p[NE];
        float mx = -3.0e38f;
#pragma unroll
        for (int e = 0; e < NE; ++e) { p[e] = acc[e] + bgate[e]; mx = fmaxf(mx, p[e]); }
        float denom = 0.f;
#pragma unroll
        for (int e = 0; e < NE; ++e) { p[e] = __expf(p[e] - mx); denom += p[e]; }
        float inv = 1.f / denom;
#pragma unroll
        for (int e = 0; e < NE; ++e) p[e] *= inv;
        int i0 = 0;
#pragma unroll
        for (int e = 1; e < NE; ++e) if (p[e] > p[i0]) i0 = e;
        int i1 = (i0 == 0) ? 1 : 0;
#pragma unroll
        for (int e = 0; e < NE; ++e) if (e != i0 && p[e] > p[i1]) i1 = e;

        int pos0 = atomicAdd(&cnt[i0], 1);
        idxbuf[i0 * L_TOK + pos0] = t;
        wbuf[i0 * L_TOK + pos0]   = p[i0];
        int pos1 = atomicAdd(&cnt[i1], 1);
        idxbuf[i1 * L_TOK + pos1] = t;
        wbuf[i1 * L_TOK + pos1]   = p[i1];
    }
}

// =====================================================================
// Kernel 3: fused up+gate GEMM + SwiGLU for one expert (gathered rows)
//   Aact[slot][:] = (Xg @ Wu + bu) * silu(Xg @ Wg + bg), bf16
//   Double-buffered LDS; A-tile via async-to-LDS, weights via reg prefetch.
// =====================================================================
__global__ __launch_bounds__(256)
void moe_upgate(const unsigned short* __restrict__ Xbf,
                const float* __restrict__ Wu, const float* __restrict__ bu,
                const float* __restrict__ Wg, const float* __restrict__ bg,
                const int* __restrict__ idxbuf, const int* __restrict__ cnt,
                unsigned short* __restrict__ Aact, int expert) {
    __shared__ int srow[BM];
    __shared__ unsigned short sA[2][BM * LDA];
    __shared__ unsigned short sU[2][BN * LDB];
    __shared__ unsigned short sG[2][BN * LDB];

    const int count = cnt[expert];
    const int m0 = blockIdx.y * BM;
    if (m0 >= count) return;
    const int n0 = blockIdx.x * BN;

    const int tid  = threadIdx.x;
    const int lane = tid & 31;
    const int wave = tid >> 5;
    const int wm = wave & 3;   // 4 waves along M
    const int wn = wave >> 2;  // 2 waves along N (each owns 64 cols)

    if (tid < BM) {
        int slot = m0 + tid;
        srow[tid] = (slot < count) ? idxbuf[expert * L_TOK + slot] : 0;
    }
    __syncthreads();

    const float* WuE = Wu + (size_t)expert * HD * INTER;
    const float* WgE = Wg + (size_t)expert * HD * INTER;

    const int arow  = tid >> 2;        // 0..63
    const int acol  = (tid & 3) * 8;   // 0,8,16,24
    const int myrow = srow[arow];

    // per-thread quad coordinates for the weight tiles (4 quads each)
    int qk[4], qn[4];
#pragma unroll
    for (int it = 0; it < 4; ++it) {
        int q = it * 256 + tid;
        qk[it] = q >> 5;
        qn[it] = (q & 31) * 4;
    }

    v8f accU[4], accG[4];
#pragma unroll
    for (int f = 0; f < 4; ++f) { accU[f] = zero8(); accG[f] = zero8(); }

    float4 uReg[4], gReg[4];

    // ---- prologue: issue k=0 tile fetches ----
    async_load_b128(&sA[0][arow * LDA + acol],
                    (unsigned)(((size_t)myrow * HD + acol) * 2), Xbf);
#pragma unroll
    for (int it = 0; it < 4; ++it) {
        size_t goff = (size_t)qk[it] * INTER + n0 + qn[it];
        uReg[it] = *(const float4*)(WuE + goff);
        gReg[it] = *(const float4*)(WgE + goff);
    }

    constexpr int KSTEPS = HD / BK;    // 64
    for (int s = 0; s < KSTEPS; ++s) {
        const int cur = s & 1;
        // store prefetched weight tile (waits its own loadcnt)
#pragma unroll
        for (int it = 0; it < 4; ++it) {
            unsigned short* du = &sU[cur][0];
            unsigned short* dg = &sG[cur][0];
            int k = qk[it], nq = qn[it];
            du[(nq + 0) * LDB + k] = f2bf(uReg[it].x);
            du[(nq + 1) * LDB + k] = f2bf(uReg[it].y);
            du[(nq + 2) * LDB + k] = f2bf(uReg[it].z);
            du[(nq + 3) * LDB + k] = f2bf(uReg[it].w);
            dg[(nq + 0) * LDB + k] = f2bf(gReg[it].x);
            dg[(nq + 1) * LDB + k] = f2bf(gReg[it].y);
            dg[(nq + 2) * LDB + k] = f2bf(gReg[it].z);
            dg[(nq + 3) * LDB + k] = f2bf(gReg[it].w);
        }
        wait_asynccnt0();          // A tile for 'cur' landed in LDS
        __syncthreads();           // publish cur buffers

        if (s + 1 < KSTEPS) {      // issue next tile fetches (overlap with WMMA)
            int kt = (s + 1) * BK;
            async_load_b128(&sA[cur ^ 1][arow * LDA + acol],
                            (unsigned)(((size_t)myrow * HD + kt + acol) * 2), Xbf);
#pragma unroll
            for (int it = 0; it < 4; ++it) {
                size_t goff = (size_t)(kt + qk[it]) * INTER + n0 + qn[it];
                uReg[it] = *(const float4*)(WuE + goff);
                gReg[it] = *(const float4*)(WgE + goff);
            }
        }

        v16bf afrag = load_a_frag(&sA[cur][0], wm * 16, lane);
#pragma unroll
        for (int f = 0; f < 4; ++f) {
            v16bf bfU = load_b_frag(&sU[cur][0], wn * 64 + f * 16, lane);
            accU[f] = __builtin_amdgcn_wmma_f32_16x16x32_bf16(
                false, afrag, false, bfU, (short)0, accU[f], false, false);
            v16bf bfG = load_b_frag(&sG[cur][0], wn * 64 + f * 16, lane);
            accG[f] = __builtin_amdgcn_wmma_f32_16x16x32_bf16(
                false, afrag, false, bfG, (short)0, accG[f], false, false);
        }
        // trailing barrier: all waves finish their WMMA reads of the cur buffers
        // before any wave begins step s+1.
        __syncthreads();
    }

    // ---- SwiGLU epilogue -> compacted bf16 activations ----
    const float* buE = bu + (size_t)expert * INTER;
    const float* bgE = bg + (size_t)expert * INTER;
    const int h  = lane >> 4;
    const int nl = lane & 15;
#pragma unroll
    for (int r = 0; r < 8; ++r) {
        int slot = m0 + wm * 16 + r + 8 * h;
        if (slot < count) {
            unsigned short* dst = Aact + (size_t)slot * INTER;
#pragma unroll
            for (int f = 0; f < 4; ++f) {
                int col = n0 + wn * 64 + f * 16 + nl;
                float u = accU[f][r] + buE[col];
                float g = accG[f][r] + bgE[col];
                float sg = g / (1.f + __expf(-g));   // silu(g)
                dst[col] = f2bf(u * sg);
            }
        }
    }
}

// =====================================================================
// Kernel 4: down-projection GEMM + weighted scatter-add for one expert
//   out[tok][:] += w * (Aact[slot] @ Wd + bd)
// =====================================================================
__global__ __launch_bounds__(256)
void moe_down(const unsigned short* __restrict__ Aact,
              const float* __restrict__ Wd, const float* __restrict__ bd,
              const int* __restrict__ idxbuf, const float* __restrict__ wbuf,
              const int* __restrict__ cnt, float* __restrict__ out, int expert) {
    __shared__ unsigned short sA[2][BM * LDA];
    __shared__ unsigned short sW[2][BN * LDB];

    const int count = cnt[expert];
    const int m0 = blockIdx.y * BM;
    if (m0 >= count) return;
    const int n0 = blockIdx.x * BN;

    const int tid  = threadIdx.x;
    const int lane = tid & 31;
    const int wave = tid >> 5;
    const int wm = wave & 3;
    const int wn = wave >> 2;

    const float* WdE = Wd + (size_t)expert * INTER * HD;

    const int arow = tid >> 2;
    const int acol = (tid & 3) * 8;

    int qk[4], qn[4];
#pragma unroll
    for (int it = 0; it < 4; ++it) {
        int q = it * 256 + tid;
        qk[it] = q >> 5;
        qn[it] = (q & 31) * 4;
    }

    v8f acc[4];
#pragma unroll
    for (int f = 0; f < 4; ++f) acc[f] = zero8();

    float4 wReg[4];

    async_load_b128(&sA[0][arow * LDA + acol],
                    (unsigned)(((size_t)(m0 + arow) * INTER + acol) * 2), Aact);
#pragma unroll
    for (int it = 0; it < 4; ++it)
        wReg[it] = *(const float4*)(WdE + (size_t)qk[it] * HD + n0 + qn[it]);

    constexpr int KSTEPS = INTER / BK;   // 32
    for (int s = 0; s < KSTEPS; ++s) {
        const int cur = s & 1;
#pragma unroll
        for (int it = 0; it < 4; ++it) {
            unsigned short* dw = &sW[cur][0];
            int k = qk[it], nq = qn[it];
            dw[(nq + 0) * LDB + k] = f2bf(wReg[it].x);
            dw[(nq + 1) * LDB + k] = f2bf(wReg[it].y);
            dw[(nq + 2) * LDB + k] = f2bf(wReg[it].z);
            dw[(nq + 3) * LDB + k] = f2bf(wReg[it].w);
        }
        wait_asynccnt0();
        __syncthreads();

        if (s + 1 < KSTEPS) {
            int kt = (s + 1) * BK;
            async_load_b128(&sA[cur ^ 1][arow * LDA + acol],
                            (unsigned)(((size_t)(m0 + arow) * INTER + kt + acol) * 2), Aact);
#pragma unroll
            for (int it = 0; it < 4; ++it)
                wReg[it] = *(const float4*)(WdE + (size_t)(kt + qk[it]) * HD + n0 + qn[it]);
        }

        v16bf afrag = load_a_frag(&sA[cur][0], wm * 16, lane);
#pragma unroll
        for (int f = 0; f < 4; ++f) {
            v16bf bf = load_b_frag(&sW[cur][0], wn * 64 + f * 16, lane);
            acc[f] = __builtin_amdgcn_wmma_f32_16x16x32_bf16(
                false, afrag, false, bf, (short)0, acc[f], false, false);
        }
        __syncthreads();
    }

    // ---- weighted scatter epilogue (experts run sequentially: no races) ----
    const float* bdE = bd + (size_t)expert * HD;
    const int h  = lane >> 4;
    const int nl = lane & 15;
#pragma unroll
    for (int r = 0; r < 8; ++r) {
        int slot = m0 + wm * 16 + r + 8 * h;
        if (slot < count) {
            int tok   = idxbuf[expert * L_TOK + slot];
            float wgt = wbuf[expert * L_TOK + slot];
            float* orow = out + (size_t)tok * HD;
#pragma unroll
            for (int f = 0; f < 4; ++f) {
                int col = n0 + wn * 64 + f * 16 + nl;
                orow[col] += wgt * (acc[f][r] + bdE[col]);
            }
        }
    }
}

// =====================================================================
// Host launcher
// =====================================================================
static inline size_t align256(size_t x) { return (x + 255) & ~(size_t)255; }

extern "C" void kernel_launch(void* const* d_in, const int* in_sizes, int n_in,
                              void* d_out, int out_size, void* d_ws, size_t ws_size,
                              hipStream_t stream) {
    (void)in_sizes; (void)n_in; (void)out_size; (void)ws_size;
    const float* x     = (const float*)d_in[0];
    const float* Wu    = (const float*)d_in[1];
    const float* bu    = (const float*)d_in[2];
    const float* Wg    = (const float*)d_in[3];
    const float* bg    = (const float*)d_in[4];
    const float* Wd    = (const float*)d_in[5];
    const float* bd    = (const float*)d_in[6];
    const float* Wgate = (const float*)d_in[7];
    const float* bgate = (const float*)d_in[8];
    float* out = (float*)d_out;

    char* ws = (char*)d_ws;
    size_t off = 0;
    unsigned short* Xbf  = (unsigned short*)(ws + off); off += align256((size_t)L_TOK * HD * 2);
    unsigned short* Aact = (unsigned short*)(ws + off); off += align256((size_t)L_TOK * INTER * 2);
    int*   idxbuf = (int*)  (ws + off); off += align256((size_t)NE * L_TOK * 4);
    float* wbuf   = (float*)(ws + off); off += align256((size_t)NE * L_TOK * 4);
    int*   cnt    = (int*)  (ws + off); off += 256;

    moe_prep<<<4096, 256, 0, stream>>>(x, Xbf, out, cnt, (size_t)L_TOK * HD);
    moe_router<<<L_TOK / 8, 256, 0, stream>>>(x, Wgate, bgate, idxbuf, wbuf, cnt);

    for (int e = 0; e < NE; ++e) {
        dim3 gA(INTER / BN, L_TOK / BM);   // 8 x 128 (early-exit past cnt[e])
        moe_upgate<<<gA, 256, 0, stream>>>(Xbf, Wu, bu, Wg, bg, idxbuf, cnt, Aact, e);
        dim3 gB(HD / BN, L_TOK / BM);      // 16 x 128
        moe_down<<<gB, 256, 0, stream>>>(Aact, Wd, bd, idxbuf, wbuf, cnt, out, e);
    }
}
